// ACTLossHead_51651276701880
// MI455X (gfx1250) — compile-verified, matrix operand order checked by reading, not run on hardware
//
#include <hip/hip_runtime.h>

// ---------------- problem constants ----------------
#define GW       40
#define GL       1600      // GW*GW
#define NB       256
#define NV       64
#define PATH_ID  9
#define START_ID 7
#define END_ID   8
#define OBST_ID  1
#define FINF     1.0e9f
#define BIGI     1600      // GRID_W*GRID_W in reference

typedef __attribute__((ext_vector_type(16))) _Float16 v16h;
typedef __attribute__((ext_vector_type(8)))  float    v8f;
typedef unsigned int u32x4 __attribute__((ext_vector_type(4)));
typedef unsigned int u32x8 __attribute__((ext_vector_type(8)));

// ---------------- CDNA5 Tensor Data Mover: 1-D tile DMA into LDS ----------------
// Builds a D# (groups 0+1, tensors up to 2D) per CDNA5 ISA
// and issues TENSOR_LOAD_TO_LDS. One issue per wave; EXEC ignored by TDM.
//   elem_code: 0=1B 1=2B 2=4B 3=8B ; nelem: elements in dim0 (tile == tensor line)
__device__ __forceinline__ void tdm_load_1d(const void* gptr, unsigned lds_off,
                                            unsigned nelem, unsigned elem_code) {
    unsigned long long ga = (unsigned long long)(uintptr_t)gptr;
    u32x4 g0;
    g0[0] = 1u;                                            // count=1, user D#
    g0[1] = lds_off;                                       // lds_addr [63:32]
    g0[2] = (unsigned)(ga & 0xFFFFFFFFu);                  // global_addr lo
    g0[3] = (unsigned)((ga >> 32) & 0x01FFFFFFu)           // global_addr hi [120:96]
          | (2u << 30);                                    // type=2 ("image") [127:126]
    u32x8 g1;
    g1[0] = (elem_code << 16);                             // wg_mask=0, data_size[17:16]
    g1[1] = (nelem & 0xFFFFu) << 16;                       // tensor_dim0 lo16 -> [31:16]
    g1[2] = (nelem >> 16) | (1u << 16);                    // td0 hi16 | tensor_dim1=1
    g1[3] = (nelem & 0xFFFFu) << 16;                       // td1 hi=0 | tile_dim0
    g1[4] = 1u;                                            // tile_dim1=1, tile_dim2=0
    g1[5] = nelem;                                         // tensor_dim0_stride lo32
    g1[6] = 0u;                                            // stride0 hi | stride1 lo
    g1[7] = 0u;                                            // stride1 hi
    asm volatile("tensor_load_to_lds %0, %1" :: "s"(g0), "s"(g1) : "memory");
}

__device__ __forceinline__ void wait_tensorcnt0() {
#if __has_builtin(__builtin_amdgcn_s_wait_tensorcnt)
    __builtin_amdgcn_s_wait_tensorcnt(0);
#else
    asm volatile("s_wait_tensorcnt 0x0" ::: "memory");
#endif
}

__device__ __forceinline__ void wait_asynccnt0() {
#if __has_builtin(__builtin_amdgcn_s_wait_asynccnt)
    __builtin_amdgcn_s_wait_asynccnt(0);
#else
    asm volatile("s_wait_asynccnt 0x0" ::: "memory");
#endif
}

// ---------------- accumulator init ----------------
__global__ void k_init(float* __restrict__ accf, unsigned int* __restrict__ accu) {
    if (threadIdx.x < 4) { accf[threadIdx.x] = 0.0f; accu[threadIdx.x] = 0u; }
}

// ---------------- phase 1: softmax path-cost + argmax (HBM-bound stream) ----------------
__global__ void k_row_softmax(const float* __restrict__ logits,
                              float* __restrict__ cost,
                              unsigned char* __restrict__ pred,
                              int nrows) {
    int r = blockIdx.x * blockDim.x + threadIdx.x;
    if (r >= nrows) return;
    const float4* row4 = (const float4*)(logits + (size_t)r * NV);
    // CDNA5 speculative prefetch (global_prefetch_b8)
    __builtin_prefetch(row4 + 16, 0, 0);

    float vals[NV];
#pragma unroll
    for (int i = 0; i < NV / 4; ++i) {
        float4 v = row4[i];
        vals[4 * i + 0] = v.x; vals[4 * i + 1] = v.y;
        vals[4 * i + 2] = v.z; vals[4 * i + 3] = v.w;
    }
    // first-occurrence argmax (jnp.argmax tie-break) + max
    float m = vals[0]; int am = 0;
#pragma unroll
    for (int k = 1; k < NV; ++k) {
        if (vals[k] > m) { m = vals[k]; am = k; }
    }
    float s = 0.0f;
#pragma unroll
    for (int k = 0; k < NV; ++k) s += expf(vals[k] - m);
    float p = expf(vals[PATH_ID] - m) / s;
    p = fmaxf(p, 1e-6f);                      // jnp.clip(path_p, 1e-6)
    cost[r] = -logf(p);
    pred[r] = (unsigned char)(am == PATH_ID); // pred_grid element
}

// ---------------- phase 2: first positions of START/END per batch ----------------
__global__ void k_find_se(const int* __restrict__ labels, int4* __restrict__ se) {
    int b = blockIdx.x;
    __shared__ int mn[2];
    if (threadIdx.x == 0) { mn[0] = BIGI; mn[1] = BIGI; }
    __syncthreads();
    const int* lb = labels + (size_t)b * GL;
    int m7 = BIGI, m8 = BIGI;
    for (int i = threadIdx.x; i < GL; i += blockDim.x) {
        int v = lb[i];
        if (v == START_ID && i < m7) m7 = i;
        if (v == END_ID   && i < m8) m8 = i;
    }
    atomicMin(&mn[0], m7);
    atomicMin(&mn[1], m8);
    __syncthreads();
    if (threadIdx.x == 0) {
        int s = mn[0], e = mn[1];
        int sr = (s < BIGI) ? s / GW : 0, sc = (s < BIGI) ? s % GW : 0;
        int er = (e < BIGI) ? e / GW : 0, ec = (e < BIGI) ? e % GW : 0;
        se[b] = make_int4(sr, sc, er, ec);
    }
}

// ---------------- phase 3: LDS-resident Jacobi relaxation + backtrace ----------------
// One workgroup per batch element. Cost tile is DMA'd into LDS by the Tensor Data
// Mover (one issue from wave 0, TENSORcnt-tracked). Early-exit on fixed point is
// bit-exact: min-plus Jacobi is monotone, so once a sweep changes nothing, the
// remaining sweeps of the reference's 1600 are identities.
__global__ void k_relax_back(const float* __restrict__ cost,
                             const int* __restrict__ labels,
                             const int4* __restrict__ se,
                             unsigned int* __restrict__ mask_cnt) {
    int b = blockIdx.x;
    __shared__ __align__(16) float cL[GL];
    __shared__ __align__(16) float dA[GL];
    __shared__ __align__(16) float dB[GL];
    __shared__ __align__(16) unsigned char road[GL];
    __shared__ __align__(16) unsigned char mk[GL];
    __shared__ int scnt;

    const float* cb = cost + (size_t)b * GL;
    const int*   lb = labels + (size_t)b * GL;
    int4 s4 = se[b];
    int sidx = s4.x * GW + s4.y;
    int eidx = s4.z * GW + s4.w;

    // TDM: wave 0 pulls the whole 6.4KB cost tile straight into LDS
    if (threadIdx.x < 32) {
        tdm_load_1d(cb, (unsigned)(uintptr_t)cL, GL, 2u /*4-byte*/);
        wait_tensorcnt0();
    }
    for (int i = threadIdx.x; i < GL; i += blockDim.x) {
        road[i] = (unsigned char)(lb[i] != OBST_ID);
        dA[i]   = FINF;
        mk[i]   = 0;
    }
    if (threadIdx.x == 0) scnt = 0;
    __syncthreads();
    if (threadIdx.x == 0) dA[sidx] = 0.0f;   // dist0.at[start].set(0)
    __syncthreads();

    float* src = dA; float* dst = dB;
    for (int it = 0; it < GL; ++it) {
        int changed = 0;
        for (int i = threadIdx.x; i < GL; i += blockDim.x) {
            float old = src[i];
            int r = i / GW, c = i % GW;
            float rt = (c < GW - 1) ? src[i + 1]  : FINF;
            float lf = (c > 0)      ? src[i - 1]  : FINF;
            float dn = (r < GW - 1) ? src[i + GW] : FINF;
            float up = (r > 0)      ? src[i - GW] : FINF;
            float nb = fminf(fminf(rt, lf), fminf(dn, up));
            float cand = nb + cL[i] + 1.0f;
            float nv = road[i] ? fminf(old, cand) : old;
            dst[i] = nv;
            changed |= (nv != old);
        }
        int any = __syncthreads_count(changed);
        float* t = src; src = dst; dst = t;
        if (any == 0) break;
    }
    // src holds converged dist

    if (threadIdx.x == 0) {
        bool done = !(src[eidx] < 0.5f * FINF);   // done = ~reach
        int r = s4.z, c = s4.w;
        const int drs[4] = {0, 0, 1, -1};
        const int dcs[4] = {1, -1, 0, 0};
        for (int it = 0; it < GL && !done; ++it) {
            mk[r * GW + c] = 1;                   // mark while active
            if (r == s4.x && c == s4.y) { done = true; break; }
            float best = 3.4e38f; int br = r, bc = c;
#pragma unroll
            for (int j = 0; j < 4; ++j) {
                int nr = r + drs[j], nc = c + dcs[j];
                bool valid = (nr >= 0) && (nr < GW) && (nc >= 0) && (nc < GW);
                int cr = nr < 0 ? 0 : (nr > GW - 1 ? GW - 1 : nr);
                int cc = nc < 0 ? 0 : (nc > GW - 1 ? GW - 1 : nc);
                float nd = valid ? src[cr * GW + cc] : FINF;
                if (nd < best) { best = nd; br = cr; bc = cc; } // first-min tie-break
            }
            r = br; c = bc;
        }
    }
    __syncthreads();

    int local = 0;
    for (int i = threadIdx.x; i < GL; i += blockDim.x) local += mk[i];
    atomicAdd(&scnt, local);
    __syncthreads();
    if (threadIdx.x == 0) atomicAdd(mask_cnt, (unsigned int)scnt);
}

// ---------------- phase 4: connected components (min-label propagation) ----------------
// Pred tile (1600B) is moved by async global->LDS copies (ASYNCcnt-tracked,
// no VGPR staging): 100 lanes x 16B.
__global__ void k_components(const unsigned char* __restrict__ pred,
                             float* __restrict__ conn_acc) {
    int b = blockIdx.x;
    __shared__ __align__(16) int lA[GL];
    __shared__ __align__(16) int lB[GL];
    __shared__ __align__(16) unsigned char pL[GL];
    __shared__ int rC, rN;

    const unsigned char* pb = pred + (size_t)b * GL;
    {
        int t = threadIdx.x;
        if (t < GL / 16) {                         // 100 lanes, 16B each
            unsigned loff = (unsigned)(uintptr_t)pL + (unsigned)t * 16u;
            unsigned goff = (unsigned)t * 16u;
            asm volatile("global_load_async_to_lds_b128 %0, %1, %2"
                         :: "v"(loff), "v"(goff), "s"(pb) : "memory");
        }
    }
    if (threadIdx.x == 0) { rC = 0; rN = 0; }
    wait_asynccnt0();
    __syncthreads();

    for (int i = threadIdx.x; i < GL; i += blockDim.x) {
        lA[i] = pL[i] ? i : BIGI;
    }
    __syncthreads();

    int* src = lA; int* dst = lB;
    for (int it = 0; it < GL; ++it) {
        int changed = 0;
        for (int i = threadIdx.x; i < GL; i += blockDim.x) {
            int old = src[i];
            int nv;
            if (pL[i]) {
                int r = i / GW, c = i % GW;
                int rt = (c < GW - 1) ? src[i + 1]  : BIGI;
                int lf = (c > 0)      ? src[i - 1]  : BIGI;
                int dn = (r < GW - 1) ? src[i + GW] : BIGI;
                int up = (r > 0)      ? src[i - GW] : BIGI;
                int nb = min(min(rt, lf), min(dn, up));
                nv = min(old, nb);
            } else {
                nv = BIGI;
            }
            dst[i] = nv;
            changed |= (nv != old);
        }
        int any = __syncthreads_count(changed);
        int* t = src; src = dst; dst = t;
        if (any == 0) break;
    }

    int lc = 0, ln = 0;
    for (int i = threadIdx.x; i < GL; i += blockDim.x) {
        if (pL[i]) { ln++; if (src[i] == i) lc++; }
    }
    atomicAdd(&rC, lc);
    atomicAdd(&rN, ln);
    __syncthreads();
    if (threadIdx.x == 0) {
        float conn = (rN > 1) ? (float)(rC - 1) : 0.0f;  // comps>=1 when npath>=1
        atomicAdd(conn_acc, conn);
    }
}

// ---------------- phase 5: WMMA exact mask counting (pc, fp, tc per batch) ----------------
// One wave handles 16 batches: D = A(16x32 of {0,1} f16) x ones(32x16) accumulated
// in f32 — exact integer counting on the matrix pipe. Lane l (l<16) supplies
// elements [c*32, c*32+16) of batch row b0+l; lane l+16 supplies [c*32+16, c*32+32),
// so each row's 32 chunk values appear exactly once (B==ones makes the intra-row
// K-placement irrelevant).
__global__ void k_wmma_counts(const unsigned char* __restrict__ pred,
                              const int* __restrict__ labels,
                              float* __restrict__ cov_acc,
                              float* __restrict__ bud_acc) {
    int lane = threadIdx.x;            // blockDim.x == 32 (one wave)
    int b0 = blockIdx.x * 16;
    int row = b0 + (lane & 15);
    int koff = (lane < 16) ? 0 : 16;
    const unsigned char* pr = pred + (size_t)row * GL + koff;
    const int*           lb = labels + (size_t)row * GL + koff;

    v16h ones;
#pragma unroll
    for (int k = 0; k < 16; ++k) ones[k] = (_Float16)1.0f;

    v8f c_pc = {}; v8f c_fp = {}; v8f c_tc = {};

    for (int ch = 0; ch < GL / 32; ++ch) {
        int base = ch * 32;
        v16h ap, af, at;
#pragma unroll
        for (int k = 0; k < 16; ++k) {
            bool pm = pr[base + k] != 0;
            bool tm = lb[base + k] == PATH_ID;
            ap[k] = pm ? (_Float16)1.0f : (_Float16)0.0f;
            at[k] = tm ? (_Float16)1.0f : (_Float16)0.0f;
            af[k] = (pm && !tm) ? (_Float16)1.0f : (_Float16)0.0f;
        }
        c_pc = __builtin_amdgcn_wmma_f32_16x16x32_f16(false, ap, false, ones,
                                                      (short)0, c_pc, false, false);
        c_fp = __builtin_amdgcn_wmma_f32_16x16x32_f16(false, af, false, ones,
                                                      (short)0, c_fp, false, false);
        c_tc = __builtin_amdgcn_wmma_f32_16x16x32_f16(false, at, false, ones,
                                                      (short)0, c_tc, false, false);
    }

    // C layout: lane<16 -> rows 0..7 in vgprs 0..7 (col = lane%16); lane>=16 -> rows 8..15.
    // Column choice irrelevant (all equal); lanes 0 and 16 cover all 16 batches.
    if ((lane & 15) == 0) {
#pragma unroll
        for (int j = 0; j < 8; ++j) {
            float pc = c_pc[j], fp = c_fp[j], tc = c_tc[j];
            float cov = fp / fmaxf(pc, 1.0f);
            float bud = fabsf(pc - tc) * (1.0f / (float)GL);
            atomicAdd(cov_acc, cov);
            atomicAdd(bud_acc, bud);
        }
    }
}

// ---------------- phase 6: combine scalar ----------------
__global__ void k_finalize(const float* __restrict__ accf,
                           const unsigned int* __restrict__ accu,
                           float* __restrict__ out) {
    // projection loss: pj == mask in {0,1}; per-element value depends only on mask
    const float C1 = 0.31326168751822286f;  // softplus(1) - 1
    const float C0 = 0.69314718055994531f;  // softplus(0)
    double BL = (double)NB * (double)GL;
    double cnt1 = (double)accu[0];
    float proj = (float)((cnt1 * (double)C1 + (BL - cnt1) * (double)C0) / BL);
    float conn = accf[0] / (float)NB;
    float cov  = accf[1] / (float)NB;
    float bud  = accf[2] / (float)NB;
    out[0] = proj + conn + cov + bud;
}

// ---------------- launcher ----------------
extern "C" void kernel_launch(void* const* d_in, const int* in_sizes, int n_in,
                              void* d_out, int out_size, void* d_ws, size_t ws_size,
                              hipStream_t stream) {
    const float* logits = (const float*)d_in[0];   // (256, 1600, 64) f32
    const int*   labels = (const int*)d_in[1];     // (256, 1600) i32
    (void)in_sizes; (void)n_in; (void)out_size; (void)ws_size;

    char* ws = (char*)d_ws;
    float*        accf = (float*)ws;                               // [0]=conn [1]=cov [2]=bud
    unsigned int* accu = (unsigned int*)(ws + 16);                 // [0]=mask1 count
    float*        cost = (float*)(ws + 256);                       // 256*1600 f32
    unsigned char* pred = (unsigned char*)(ws + 256 + (size_t)NB * GL * 4);
    int4*         se   = (int4*)(ws + 256 + (size_t)NB * GL * 4 + (size_t)NB * GL);

    k_init<<<1, 32, 0, stream>>>(accf, accu);
    k_row_softmax<<<(NB * GL + 255) / 256, 256, 0, stream>>>(logits, cost, pred, NB * GL);
    k_find_se<<<NB, 256, 0, stream>>>(labels, se);
    k_relax_back<<<NB, 256, 0, stream>>>(cost, labels, se, accu);
    k_components<<<NB, 256, 0, stream>>>(pred, &accf[0]);
    k_wmma_counts<<<NB / 16, 32, 0, stream>>>(pred, labels, &accf[1], &accf[2]);
    k_finalize<<<1, 1, 0, stream>>>(accf, accu, (float*)d_out);
}
